// SubstructureAwareGNN_17514876634163
// MI455X (gfx1250) — matrix operand order repaired and checked by reading, not compiled, not run on hardware
//
#include <hip/hip_runtime.h>
#include <hip/hip_bf16.h>

// Problem constants (match reference)
#define GNN_N   4096
#define GNN_E   131072
#define GNN_IN  256
#define GNN_HID 512
#define GNN_OUT 64
#define NW      (GNN_N / 32)   // 128 dwords per bitset row

typedef __attribute__((ext_vector_type(16))) __bf16 v16bf;
typedef __attribute__((ext_vector_type(8)))  float  v8f;

// branchless round-to-nearest-even f32 -> bf16, packed pair into one dword
__device__ __forceinline__ unsigned pack_bf16(float x, float y) {
    unsigned ux = __float_as_uint(x);
    ux = (ux + 0x7fffu + ((ux >> 16) & 1u)) >> 16;
    unsigned uy = __float_as_uint(y);
    uy = (uy + 0x7fffu + ((uy >> 16) & 1u)) >> 16;
    return ux | (uy << 16);
}

// ---------------------------------------------------------------------------
// 1) Build in-neighbor bitsets: inmask[t] bit s set iff edge s->t exists.
// ---------------------------------------------------------------------------
__global__ __launch_bounds__(256)
void k_build_inmask(const int* __restrict__ src, const int* __restrict__ dst,
                    unsigned* __restrict__ inmask) {
    int e = blockIdx.x * 256 + threadIdx.x;
    if (e >= GNN_E) return;
    int s = src[e], t = dst[e];
    atomicOr(&inmask[(size_t)t * NW + (s >> 5)], 1u << (s & 31));
}

// ---------------------------------------------------------------------------
// 2) 2-hop reachability: reach(t) = {t} | inmask(t) | U_{m in inmask(t)} inmask(m)
// ---------------------------------------------------------------------------
__global__ __launch_bounds__(128)
void k_reach(const unsigned* __restrict__ inmask,
             unsigned* __restrict__ reachbits, float* __restrict__ invcnt) {
    __shared__ unsigned orig[NW];
    __shared__ int red[NW];
    int t = blockIdx.x, tid = threadIdx.x;
    unsigned a = inmask[(size_t)t * NW + tid];
    orig[tid] = a;
    if (tid == (t >> 5)) a |= 1u << (t & 31);   // self bit
    __syncthreads();
    for (int j = 0; j < NW; ++j) {
        unsigned w = orig[j];                   // uniform LDS broadcast
        while (w) {
            int m = j * 32 + __builtin_ctz(w);
            w &= w - 1;
            a |= inmask[(size_t)m * NW + tid];  // coalesced 512B row slice
        }
    }
    reachbits[(size_t)t * NW + tid] = a;
    red[tid] = __popc(a);
    __syncthreads();
    for (int s = 64; s > 0; s >>= 1) {
        if (tid < s) red[tid] += red[tid + s];
        __syncthreads();
    }
    if (tid == 0) invcnt[t] = 1.0f / (float)red[0];   // count >= 1 (self)
}

// ---------------------------------------------------------------------------
// 3) cut subgraph: keep a deterministic exact half of the edges, segment-mean
//    of x[dst] over kept out-edges per src, fallback to x.
// ---------------------------------------------------------------------------
__global__ __launch_bounds__(256)
void k_cut_accum(const int* __restrict__ src, const int* __restrict__ dst,
                 const float* __restrict__ x,
                 float* __restrict__ sums, float* __restrict__ cnt) {
    int gid = blockIdx.x * 256 + threadIdx.x;     // E * 32 threads
    int e = gid >> 5, c = gid & 31;
    if (e >= GNN_E) return;
    if (!(e & 1)) return;                         // keep exactly E/2 edges
    int s = src[e], t = dst[e];
    int f0 = c * 8;
    #pragma unroll
    for (int f = 0; f < 8; ++f)
        atomicAdd(&sums[(size_t)s * GNN_IN + f0 + f], x[(size_t)t * GNN_IN + f0 + f]);
    if (c == 0) atomicAdd(&cnt[s], 1.0f);
}

__global__ __launch_bounds__(256)
void k_cut_finalize(const float* __restrict__ x, const float* __restrict__ cnt,
                    float* __restrict__ sums /* in-place -> cut_feat */) {
    int gid = blockIdx.x * 256 + threadIdx.x;     // N * IN threads
    int n = gid >> 8;
    float c = cnt[n];
    sums[gid] = (c > 0.0f) ? sums[gid] / c : x[gid];
}

// ---------------------------------------------------------------------------
// 4) bf16-WMMA GEMM:  C[M,NC] (=|+=) rowscale * (A[M,K] @ W[K,NC] + bias)
//    Template BITS selects A source (f32 array vs reachability bitset rows),
//    so both paths are branch-free.
//    Block = 128 threads = 4 waves; block tile 64x16; wave tile 16x16; BK=32.
//
//    LDS layout (dword granularity, row stride 20 dwords = 80B, 16B aligned):
//      AsU : 64 rows x 16 dwords, dword d of a row = bf16 pair (k=2d, 2d+1)
//      BsU : 16 cols x 16 dwords (B tile TRANSPOSED), dword d = (k=2d, 2d+1)
//    With the ISA 7.12.2 wave32 fragment layouts, each lane's A fragment is
//    two 16B-aligned 4-dword runs and its B fragment the same ->
//    wide ds_load_b128s instead of 32x ds_load_u16.
// ---------------------------------------------------------------------------
#define LDS_STRIDE 20

template <bool BITS>
__global__ __launch_bounds__(128)
void k_gemm_wmma(const float* __restrict__ A, const unsigned* __restrict__ Abits,
                 const float* __restrict__ rowscale,
                 const float* __restrict__ W, const float* __restrict__ bias,
                 float* __restrict__ C, int M, int K, int NC, int accum) {
    __shared__ unsigned AsU[64 * LDS_STRIDE];
    __shared__ unsigned BsU[16 * LDS_STRIDE];

    const int tid  = threadIdx.x;
    const int lane = tid & 31;
    const int wv   = tid >> 5;
    const int m0   = blockIdx.y * 64;
    const int n0   = blockIdx.x * 16;
    const int nwA  = K >> 5;

    v8f acc = {0.f, 0.f, 0.f, 0.f, 0.f, 0.f, 0.f, 0.f};

    for (int k0 = 0; k0 < K; k0 += 32) {
        // ---- stage A tile (64 rows x 32 k) ----
        if (BITS) {
            // two threads per row; one bitset dword covers the whole k-slice
            int r = tid >> 1, half = tid & 1;
            unsigned w = Abits[(size_t)(m0 + r) * nwA + (k0 >> 5)] >> (half * 16);
            unsigned* drow = &AsU[r * LDS_STRIDE + half * 8];
            #pragma unroll
            for (int j = 0; j < 8; ++j) {
                unsigned b0 = (w >> (2 * j)) & 1u;
                unsigned b1 = (w >> (2 * j + 1)) & 1u;
                drow[j] = (b0 * 0x3F80u) | ((b1 * 0x3F80u) << 16);
            }
        } else {
            // 8 x float2 per thread, k-adjacent pair -> one packed u32 store
            #pragma unroll
            for (int i = 0; i < 8; ++i) {
                int idx2 = i * 256 + tid * 2;
                int r = idx2 >> 5, c = idx2 & 31;
                const float* ap = &A[(size_t)(m0 + r) * K + (k0 + c)];
                float2 f = *(const float2*)ap;
                AsU[r * LDS_STRIDE + (c >> 1)] = pack_bf16(f.x, f.y);
            }
        }
        // ---- stage B tile (32 k x 16 n), stored transposed ----
        #pragma unroll
        for (int i = 0; i < 2; ++i) {
            int pidx = i * 128 + tid;          // 256 k-pairs total
            int r2 = (pidx >> 4) * 2, c = pidx & 15;
            float w0 = W[(size_t)(k0 + r2) * NC + (n0 + c)];
            float w1 = W[(size_t)(k0 + r2 + 1) * NC + (n0 + c)];
            BsU[c * LDS_STRIDE + (r2 >> 1)] = pack_bf16(w0, w1);
        }
        if (k0 + 32 < K) {
            __builtin_prefetch(&W[(size_t)(k0 + 32) * NC + n0], 0, 1);
            if (!BITS)
                __builtin_prefetch(&A[(size_t)(m0 + (tid >> 1)) * K + k0 + 32], 0, 1);
        }
        __syncthreads();

        // ---- fragments (ISA 7.12.2 wave32 layouts), wide dword LDS loads ----
        const int mrow = lane & 15;       // A row / B column
        const int kh   = lane >> 4;       // K-half selector
        union { v16bf v; unsigned u[8]; } a, b;
        const unsigned* arow = &AsU[(wv * 16 + mrow) * LDS_STRIDE];
        #pragma unroll
        for (int j = 0; j < 4; ++j) {
            a.u[j]     = arow[kh * 4 + j];        // K = kh*8 + 2j, +1
            a.u[4 + j] = arow[8 + kh * 4 + j];    // K = 16 + kh*8 + 2j, +1
        }
        const unsigned* brow = &BsU[mrow * LDS_STRIDE];
        #pragma unroll
        for (int j = 0; j < 4; ++j) {
            b.u[j]     = brow[kh * 4 + j];        // K = kh*8 + 2j, +1
            b.u[4 + j] = brow[8 + kh * 4 + j];    // K = 16 + kh*8 + 2j, +1
        }

        acc = __builtin_amdgcn_wmma_f32_16x16x32_bf16(
                  false, a.v, false, b.v, (short)0, acc, false, false);
        __syncthreads();
    }

    // epilogue: C 16x16 f32 layout — VGPR r: lanes0-15 M=r, lanes16-31 M=r+8
    const int col   = n0 + (lane & 15);
    const int rbase = m0 + wv * 16 + ((lane >> 4) << 3);
    #pragma unroll
    for (int r = 0; r < 8; ++r) {
        int row = rbase + r;
        float v = acc[r];
        if (bias)     v += bias[col];
        if (rowscale) v *= rowscale[row];
        size_t o = (size_t)row * NC + col;
        if (accum) C[o] += v; else C[o] = v;
    }
}

// ---------------------------------------------------------------------------
// 5) Edge scatter-add: agg[dst] += h[src]  (HID features, 8 per thread)
// ---------------------------------------------------------------------------
__global__ __launch_bounds__(256)
void k_scatter(const int* __restrict__ src, const int* __restrict__ dst,
               const float* __restrict__ h, float* __restrict__ agg) {
    int gid = blockIdx.x * 256 + threadIdx.x;   // E * 64 threads
    int e = gid >> 6, c = gid & 63;
    if (e >= GNN_E) return;
    int s = src[e], t = dst[e];
    int f0 = c * 8;
    #pragma unroll
    for (int f = 0; f < 8; ++f)
        atomicAdd(&agg[(size_t)t * GNN_HID + f0 + f], h[(size_t)s * GNN_HID + f0 + f]);
}

__global__ __launch_bounds__(256)
void k_relu(float* __restrict__ p, int n) {
    int gid = blockIdx.x * 256 + threadIdx.x;
    if (gid < n) p[gid] = fmaxf(p[gid], 0.0f);
}

// ---------------------------------------------------------------------------
// 6) log_softmax over OUT=64 cols: one wave32 per row, 2 cols/lane
// ---------------------------------------------------------------------------
__global__ __launch_bounds__(32)
void k_log_softmax(float* __restrict__ out) {
    int r = blockIdx.x, l = threadIdx.x;
    float v0 = out[(size_t)r * GNN_OUT + l];
    float v1 = out[(size_t)r * GNN_OUT + 32 + l];
    float m = fmaxf(v0, v1);
    #pragma unroll
    for (int s = 16; s > 0; s >>= 1) m = fmaxf(m, __shfl_xor(m, s, 32));
    float sum = __expf(v0 - m) + __expf(v1 - m);
    #pragma unroll
    for (int s = 16; s > 0; s >>= 1) sum += __shfl_xor(sum, s, 32);
    float lse = m + __logf(sum);
    out[(size_t)r * GNN_OUT + l]      = v0 - lse;
    out[(size_t)r * GNN_OUT + 32 + l] = v1 - lse;
}

// ---------------------------------------------------------------------------
extern "C" void kernel_launch(void* const* d_in, const int* in_sizes, int n_in,
                              void* d_out, int out_size, void* d_ws, size_t ws_size,
                              hipStream_t stream) {
    const float* x     = (const float*)d_in[0];
    const int*   eidx  = (const int*)  d_in[1];
    const float* egoW  = (const float*)d_in[2];
    const float* egob  = (const float*)d_in[3];
    const float* cutW  = (const float*)d_in[4];
    const float* cutb  = (const float*)d_in[5];
    const float* globW = (const float*)d_in[6];
    const float* globb = (const float*)d_in[7];
    const float* fcW   = (const float*)d_in[8];
    const float* fcb   = (const float*)d_in[9];
    const int* src = eidx;
    const int* dst = eidx + GNN_E;
    float* out = (float*)d_out;

    // workspace carve-out (256B aligned)
    char* base = (char*)d_ws;
    size_t off = 0;
    auto carve = [&](size_t bytes) -> char* {
        char* p = base + off;
        off = (off + bytes + 255) & ~(size_t)255;
        return p;
    };
    unsigned* inmask    = (unsigned*)carve((size_t)GNN_N * NW * 4);       // 2 MB
    unsigned* reachbits = (unsigned*)carve((size_t)GNN_N * NW * 4);       // 2 MB
    float*    invcnt    = (float*)   carve((size_t)GNN_N * 4);
    float*    ego_feat  = (float*)   carve((size_t)GNN_N * GNN_IN * 4);   // 4 MB
    float*    cut_feat  = (float*)   carve((size_t)GNN_N * GNN_IN * 4);   // 4 MB (sums, finalized in place)
    float*    cut_cnt   = (float*)   carve((size_t)GNN_N * 4);
    float*    h_ego     = (float*)   carve((size_t)GNN_N * GNN_HID * 4);  // 8 MB
    float*    h_cut     = (float*)   carve((size_t)GNN_N * GNN_HID * 4);  // 8 MB
    float*    ego_enc   = (float*)   carve((size_t)GNN_N * GNN_HID * 4);  // 8 MB
    float*    cut_enc   = (float*)   carve((size_t)GNN_N * GNN_HID * 4);  // 8 MB
    float*    glob_enc  = (float*)   carve((size_t)GNN_N * GNN_HID * 4);  // 8 MB

    // zero accumulation targets (graph-capture-safe async memsets, every call)
    hipMemsetAsync(inmask,   0, (size_t)GNN_N * NW * 4, stream);
    hipMemsetAsync(cut_feat, 0, (size_t)GNN_N * GNN_IN * 4, stream);
    hipMemsetAsync(cut_cnt,  0, (size_t)GNN_N * 4, stream);
    hipMemsetAsync(ego_enc,  0, (size_t)GNN_N * GNN_HID * 4, stream);
    hipMemsetAsync(cut_enc,  0, (size_t)GNN_N * GNN_HID * 4, stream);

    // 1-2) reachability bitsets
    k_build_inmask<<<GNN_E / 256, 256, 0, stream>>>(src, dst, inmask);
    k_reach<<<GNN_N, 128, 0, stream>>>(inmask, reachbits, invcnt);

    // 3) cut subgraph features
    k_cut_accum<<<(GNN_E * 32) / 256, 256, 0, stream>>>(src, dst, x, cut_feat, cut_cnt);
    k_cut_finalize<<<(GNN_N * GNN_IN) / 256, 256, 0, stream>>>(x, cut_cnt, cut_feat);

    // 4) GEMMs via WMMA
    // ego_feat = (reach bits) @ x * invcnt   [M=4096, K=4096, NC=256]
    k_gemm_wmma<true><<<dim3(GNN_IN / 16, GNN_N / 64), 128, 0, stream>>>(
        nullptr, reachbits, invcnt, x, nullptr, ego_feat, GNN_N, GNN_N, GNN_IN, 0);
    // h_ego = ego_feat @ ego_W + ego_b       [4096, 256, 512]
    k_gemm_wmma<false><<<dim3(GNN_HID / 16, GNN_N / 64), 128, 0, stream>>>(
        ego_feat, nullptr, nullptr, egoW, egob, h_ego, GNN_N, GNN_IN, GNN_HID, 0);
    // h_cut = cut_feat @ cut_W + cut_b
    k_gemm_wmma<false><<<dim3(GNN_HID / 16, GNN_N / 64), 128, 0, stream>>>(
        cut_feat, nullptr, nullptr, cutW, cutb, h_cut, GNN_N, GNN_IN, GNN_HID, 0);
    // glob_enc = x @ glob_W + glob_b
    k_gemm_wmma<false><<<dim3(GNN_HID / 16, GNN_N / 64), 128, 0, stream>>>(
        x, nullptr, nullptr, globW, globb, glob_enc, GNN_N, GNN_IN, GNN_HID, 0);

    // 5) message passing scatter + relu
    k_scatter<<<(GNN_E * 64) / 256, 256, 0, stream>>>(src, dst, h_ego, ego_enc);
    k_scatter<<<(GNN_E * 64) / 256, 256, 0, stream>>>(src, dst, h_cut, cut_enc);
    k_relu<<<(GNN_N * GNN_HID) / 256, 256, 0, stream>>>(ego_enc, GNN_N * GNN_HID);
    k_relu<<<(GNN_N * GNN_HID) / 256, 256, 0, stream>>>(cut_enc, GNN_N * GNN_HID);

    // 6) fc: out = [ego_enc | cut_enc | glob_enc] @ fc_W + fc_b  (3 accumulating GEMMs)
    k_gemm_wmma<false><<<dim3(GNN_OUT / 16, GNN_N / 64), 128, 0, stream>>>(
        ego_enc, nullptr, nullptr, fcW, fcb, out, GNN_N, GNN_HID, GNN_OUT, 0);
    k_gemm_wmma<false><<<dim3(GNN_OUT / 16, GNN_N / 64), 128, 0, stream>>>(
        cut_enc, nullptr, nullptr, fcW + (size_t)GNN_HID * GNN_OUT, nullptr, out,
        GNN_N, GNN_HID, GNN_OUT, 1);
    k_gemm_wmma<false><<<dim3(GNN_OUT / 16, GNN_N / 64), 128, 0, stream>>>(
        glob_enc, nullptr, nullptr, fcW + (size_t)(2 * GNN_HID) * GNN_OUT, nullptr, out,
        GNN_N, GNN_HID, GNN_OUT, 1);

    // 7) log_softmax rows
    k_log_softmax<<<GNN_N, 32, 0, stream>>>(out);
}
